// VisualIMURecurrentFusionBlock_65352222376601
// MI455X (gfx1250) — matrix operand order repaired and checked by reading, not compiled, not run on hardware
//
#include <hip/hip_runtime.h>
#include <stdint.h>

#define T_STEPS 512
#define BATCH   256
#define VIS_D   256
#define IMU_D   64
#define TRAJ_D  64
#define FUSE    384
#define HID     256
#define OUT_D   128
#define M_ROWS  (T_STEPS * BATCH)   // 131072

typedef __attribute__((ext_vector_type(16))) __bf16 v16bf;
typedef __attribute__((ext_vector_type(8)))  float  v8f;
typedef int v4i __attribute__((vector_size(16)));   // matches async-builtin param pointee

union Frag { v16bf v; unsigned int u[8]; };

__device__ __forceinline__ unsigned short f32_bf16(float f) {
  unsigned int x = __float_as_uint(f);
  unsigned int r = x + 0x7FFFu + ((x >> 16) & 1u);   // round-to-nearest-even
  return (unsigned short)(r >> 16);
}

// A fragment (16x32 bf16): lane l -> row M = l%16, group g = l/16.
// VGPR v, half j: K = (v&3)*2 + g*8 + (v>>2)*16 + j   (ISA 7.12.2 table)
__device__ __forceinline__ int a_kk(int v, int g) {
  return ((v & 3) << 1) + (g << 3) + ((v >> 2) << 4);
}
// B fragment (32x16 bf16): lane l -> col N = l%16, group g = l/16.
// VGPR v, half j: K = (v&3)*2 + g*8 + (v>>2)*16 + j   (same K order as A)
__device__ __forceinline__ int b_kk(int v, int g) {
  return ((v & 3) << 1) + (g << 3) + ((v >> 2) << 4);
}

// ---- async global->LDS (ASYNCcnt) path, with safe fallback ----------------
#if defined(__has_builtin)
#if __has_builtin(__builtin_amdgcn_global_load_async_to_lds_b128)
#define HAVE_ASYNC_LDS 1
#endif
#endif

__device__ __forceinline__ void copy16_g2l(const unsigned short* g, unsigned short* l) {
#ifdef HAVE_ASYNC_LDS
  __builtin_amdgcn_global_load_async_to_lds_b128(
      (v4i*)g, (v4i*)l, /*offset=*/0, /*cpol=*/0);
#else
  *reinterpret_cast<uint4*>(l) = *reinterpret_cast<const uint4*>(g);
#endif
}

__device__ __forceinline__ void async_join() {
#ifdef HAVE_ASYNC_LDS
#if __has_builtin(__builtin_amdgcn_s_wait_asynccnt)
  __builtin_amdgcn_s_wait_asynccnt(0);
#else
  asm volatile("s_wait_asynccnt 0x0" ::: "memory");
#endif
#endif
}

// ---------------------------------------------------------------------------
// Kernel 0: pre-pack weights to bf16, transposed to [N][K] (K contiguous)
// ---------------------------------------------------------------------------
__global__ __launch_bounds__(256) void pack_weights_kernel(
    const float* __restrict__ W_in, const float* __restrict__ W_head,
    unsigned short* __restrict__ W_in_t, unsigned short* __restrict__ W_head_t)
{
  const int idx = blockIdx.x * 256 + threadIdx.x;
  if (idx < FUSE * HID) {                 // W_in_t[n*FUSE + k] = bf16(W_in[k*HID + n])
    int n = idx / FUSE, k = idx - n * FUSE;
    W_in_t[idx] = f32_bf16(W_in[(size_t)k * HID + n]);
  }
  if (idx < HID * OUT_D) {                // W_head_t[n*HID + k] = bf16(W_head[k*OUT_D + n])
    int n = idx / HID, k = idx - n * HID;
    W_head_t[idx] = f32_bf16(W_head[(size_t)k * OUT_D + n]);
  }
}

// ---------------------------------------------------------------------------
// Kernel 1: x_proj = concat(vis,imu,traj) @ W_in + b_in
// block: 256 threads = 8 waves; block tile 64(M) x 256(N); wave tile 32x64
// ---------------------------------------------------------------------------
#define LDA 40   // ushort stride, 80B: multiple of 16B, conflict-free gathers
#define LDB 40

__global__ __launch_bounds__(256) void inproj_wmma_kernel(
    const float* __restrict__ vis, const float* __restrict__ imu,
    const float* __restrict__ traj, const unsigned short* __restrict__ W_in_t,
    const float* __restrict__ b_in, float* __restrict__ x_proj)
{
  __shared__ unsigned short As[64 * LDA];     // 64 rows x 32 K (bf16)
  __shared__ unsigned short Bs[HID * LDB];    // 256 N x 32 K (bf16, K contiguous)

  const int tid  = threadIdx.x;
  const int lane = tid & 31;
  const int wv   = tid >> 5;
  const int wm   = wv & 1;        // 2 row-groups of 32
  const int wn   = wv >> 1;       // 4 col-groups of 64
  const int rowBlk = blockIdx.x * 64;
  const int g  = lane >> 4;
  const int lm = lane & 15;

  v8f zero = {0.f,0.f,0.f,0.f,0.f,0.f,0.f,0.f};
  v8f c[2][4];
#pragma unroll
  for (int i = 0; i < 2; ++i)
#pragma unroll
    for (int j = 0; j < 4; ++j) c[i][j] = zero;

  for (int kt = 0; kt < FUSE / 32; ++kt) {
    const int k0 = kt * 32;
    // fused-concat segment select (32-wide K tiles never straddle a boundary)
    const float* src; int scol, sstride;
    if (k0 < VIS_D)              { src = vis;  scol = k0;                 sstride = VIS_D;  }
    else if (k0 < VIS_D + IMU_D) { src = imu;  scol = k0 - VIS_D;         sstride = IMU_D;  }
    else                         { src = traj; scol = k0 - VIS_D - IMU_D; sstride = TRAJ_D; }

    __syncthreads();
    { // stage B: async bf16 copy from pre-packed W_in_t (4 x 16B per thread)
      const unsigned short* s = W_in_t + (size_t)tid * FUSE + k0;
      unsigned short* d = &Bs[tid * LDB];
#pragma unroll
      for (int cb = 0; cb < 4; ++cb) copy16_g2l(s + cb * 8, d + cb * 8);
    }
    { // stage A: 64 x 32 f32 -> bf16, coalesced float4 loads
      int r  = tid >> 3;          // 0..31
      int cc = (tid & 7) << 2;    // 0..28 step 4
#pragma unroll
      for (int p = 0; p < 2; ++p) {
        int rr = r + p * 32;
        float4 f = *reinterpret_cast<const float4*>(
            src + (size_t)(rowBlk + rr) * sstride + scol + cc);
        unsigned short* d = &As[rr * LDA + cc];
        d[0] = f32_bf16(f.x); d[1] = f32_bf16(f.y);
        d[2] = f32_bf16(f.z); d[3] = f32_bf16(f.w);
      }
    }
    async_join();
    __syncthreads();

    Frag a[2], b[4];
#pragma unroll
    for (int mf = 0; mf < 2; ++mf) {
      int r = wm * 32 + mf * 16 + lm;
#pragma unroll
      for (int v = 0; v < 8; ++v)
        a[mf].u[v] = *reinterpret_cast<const unsigned int*>(&As[r * LDA + a_kk(v, g)]);
    }
#pragma unroll
    for (int nf = 0; nf < 4; ++nf) {
      int n = wn * 64 + nf * 16 + lm;
#pragma unroll
      for (int v = 0; v < 8; ++v)
        b[nf].u[v] = *reinterpret_cast<const unsigned int*>(&Bs[n * LDB + b_kk(v, g)]);
    }
#pragma unroll
    for (int mf = 0; mf < 2; ++mf)
#pragma unroll
      for (int nf = 0; nf < 4; ++nf)
        c[mf][nf] = __builtin_amdgcn_wmma_f32_16x16x32_bf16(
            false, a[mf].v, false, b[nf].v, (short)0, c[mf][nf], false, false);
  }

  // D layout: VGPR v -> M = base + v + 8*g, lane%16 -> N
#pragma unroll
  for (int mf = 0; mf < 2; ++mf) {
    int rbase = rowBlk + wm * 32 + mf * 16 + g * 8;
#pragma unroll
    for (int nf = 0; nf < 4; ++nf) {
      int col = wn * 64 + nf * 16 + lm;
      float bias = b_in[col];
#pragma unroll
      for (int v = 0; v < 8; ++v)
        x_proj[(size_t)(rbase + v) * HID + col] = c[mf][nf][v] + bias;
    }
  }
}

// ---------------------------------------------------------------------------
// Kernel 2: recurrent LIF scan (sequential in T, parallel over B*HID)
// ---------------------------------------------------------------------------
__global__ __launch_bounds__(256) void lif_scan_kernel(
    const float* __restrict__ x_proj, const float* __restrict__ w_rec,
    unsigned short* __restrict__ spikes)
{
  const int idx = blockIdx.x * 256 + threadIdx.x;   // b*HID + h
  const float wr = w_rec[idx & (HID - 1)];
  const int stride = BATCH * HID;                   // 65536
  float V = 0.f, S = 0.f;
  for (int t = 0; t < T_STEPS; ++t) {
    float x = x_proj[(size_t)t * stride + idx];
    V = 0.9f * V + x + wr * S;                      // BETA leak + input + recurrent
    float Sn = (V - 1.0f) > 0.0f ? 1.0f : 0.0f;     // Heaviside at THRESHOLD=1
    V -= Sn;                                        // soft reset
    S = Sn;
    spikes[(size_t)t * stride + idx] = (Sn > 0.f) ? (unsigned short)0x3F80 : (unsigned short)0;
  }
}

// ---------------------------------------------------------------------------
// Kernel 3: logits = spikes(bf16) @ W_head + b_head
// block tile 128(M) x 128(N); 8 waves; wave tile 32x64; K = 256
// ---------------------------------------------------------------------------
__global__ __launch_bounds__(256) void head_wmma_kernel(
    const unsigned short* __restrict__ spikes, const unsigned short* __restrict__ W_head_t,
    const float* __restrict__ b_head, float* __restrict__ logits)
{
  __shared__ unsigned short As[128 * LDA];    // 128 rows x 32 K
  __shared__ unsigned short Bs[OUT_D * LDB];  // 128 N x 32 K

  const int tid  = threadIdx.x;
  const int lane = tid & 31;
  const int wv   = tid >> 5;
  const int wm   = wv & 3;        // 4 row-groups of 32
  const int wn   = wv >> 2;       // 2 col-groups of 64
  const int rowBlk = blockIdx.x * 128;
  const int g  = lane >> 4;
  const int lm = lane & 15;

  v8f zero = {0.f,0.f,0.f,0.f,0.f,0.f,0.f,0.f};
  v8f c[2][4];
#pragma unroll
  for (int i = 0; i < 2; ++i)
#pragma unroll
    for (int j = 0; j < 4; ++j) c[i][j] = zero;

  for (int kt = 0; kt < HID / 32; ++kt) {
    const int k0 = kt * 32;
    __syncthreads();
    { // stage A: async bf16 copy, 2 x 16B per thread
      int r    = tid >> 2;          // 0..63
      int part = (tid & 3) << 3;    // 0,8,16,24
#pragma unroll
      for (int p = 0; p < 2; ++p) {
        int rr = r + p * 64;
        copy16_g2l(spikes + (size_t)(rowBlk + rr) * HID + k0 + part,
                   &As[rr * LDA + part]);
      }
    }
    { // stage B: async bf16 copy from pre-packed W_head_t (2 x 16B per thread)
      int n    = tid & (OUT_D - 1);
      int half = (tid >> 7) << 4;   // 0 or 16
      const unsigned short* s = W_head_t + (size_t)n * HID + k0 + half;
      unsigned short* d = &Bs[n * LDB + half];
      copy16_g2l(s, d);
      copy16_g2l(s + 8, d + 8);
    }
    async_join();
    __syncthreads();

    Frag a[2], b[4];
#pragma unroll
    for (int mf = 0; mf < 2; ++mf) {
      int r = wm * 32 + mf * 16 + lm;
#pragma unroll
      for (int v = 0; v < 8; ++v)
        a[mf].u[v] = *reinterpret_cast<const unsigned int*>(&As[r * LDA + a_kk(v, g)]);
    }
#pragma unroll
    for (int nf = 0; nf < 4; ++nf) {
      int n = wn * 64 + nf * 16 + lm;
#pragma unroll
      for (int v = 0; v < 8; ++v)
        b[nf].u[v] = *reinterpret_cast<const unsigned int*>(&Bs[n * LDB + b_kk(v, g)]);
    }
#pragma unroll
    for (int mf = 0; mf < 2; ++mf)
#pragma unroll
      for (int nf = 0; nf < 4; ++nf)
        c[mf][nf] = __builtin_amdgcn_wmma_f32_16x16x32_bf16(
            false, a[mf].v, false, b[nf].v, (short)0, c[mf][nf], false, false);
  }

#pragma unroll
  for (int mf = 0; mf < 2; ++mf) {
    int rbase = rowBlk + wm * 32 + mf * 16 + g * 8;
#pragma unroll
    for (int nf = 0; nf < 4; ++nf) {
      int col = wn * 64 + nf * 16 + lm;
      float bias = b_head[col];
#pragma unroll
      for (int v = 0; v < 8; ++v)
        logits[(size_t)(rbase + v) * OUT_D + col] = c[mf][nf][v] + bias;
    }
  }
}

// ---------------------------------------------------------------------------
// Kernel 4: out[b,o] = mean over t of logits[t,b,o]
// ---------------------------------------------------------------------------
__global__ __launch_bounds__(256) void mean_kernel(
    const float* __restrict__ logits, float* __restrict__ out)
{
  const int idx = blockIdx.x * 256 + threadIdx.x;   // b*OUT + o
  const int stride = BATCH * OUT_D;                 // 32768
  float s = 0.f;
  for (int t = 0; t < T_STEPS; ++t)
    s += logits[(size_t)t * stride + idx];
  out[idx] = s * (1.0f / T_STEPS);
}

// ---------------------------------------------------------------------------
extern "C" void kernel_launch(void* const* d_in, const int* in_sizes, int n_in,
                              void* d_out, int out_size, void* d_ws, size_t ws_size,
                              hipStream_t stream)
{
  (void)in_sizes; (void)n_in; (void)out_size; (void)ws_size;

  const float* vis    = (const float*)d_in[0];
  const float* imu    = (const float*)d_in[1];
  const float* traj   = (const float*)d_in[2];
  const float* W_in   = (const float*)d_in[3];
  const float* b_in   = (const float*)d_in[4];
  const float* w_rec  = (const float*)d_in[5];
  const float* W_head = (const float*)d_in[6];
  const float* b_head = (const float*)d_in[7];

  float* out    = (float*)d_out;                         // [B, OUT]
  float* logits = out + (size_t)BATCH * OUT_D;           // [T, B, OUT]

  char* ws = (char*)d_ws;
  float* x_proj = (float*)ws;                                     // M*HID f32  (128 MiB)
  ws += (size_t)M_ROWS * HID * sizeof(float);
  unsigned short* spikes = (unsigned short*)ws;                   // M*HID bf16 (64 MiB)
  ws += (size_t)M_ROWS * HID * sizeof(unsigned short);
  unsigned short* W_in_t = (unsigned short*)ws;                   // [HID][FUSE] bf16
  ws += (size_t)HID * FUSE * sizeof(unsigned short);
  unsigned short* W_head_t = (unsigned short*)ws;                 // [OUT][HID] bf16

  pack_weights_kernel<<<(FUSE * HID + 255) / 256, 256, 0, stream>>>(W_in, W_head, W_in_t, W_head_t);
  inproj_wmma_kernel<<<M_ROWS / 64, 256, 0, stream>>>(vis, imu, traj, W_in_t, b_in, x_proj);
  lif_scan_kernel<<<(BATCH * HID) / 256, 256, 0, stream>>>(x_proj, w_rec, spikes);
  head_wmma_kernel<<<M_ROWS / 128, 256, 0, stream>>>(spikes, W_head_t, b_head, logits);
  mean_kernel<<<(BATCH * OUT_D) / 256, 256, 0, stream>>>(logits, out);
}
